// GAT_layer_68272800137712
// MI455X (gfx1250) — matrix operand order repaired and checked by reading, not compiled, not run on hardware
//
#include <hip/hip_runtime.h>

// ---------------- problem constants ----------------
#define BDIM   4096
#define DMODEL 512
#define NHEADS 8
#define DHEAD  64
#define LN_EPS 1e-5f
#define NEG_SLOPE 0.2f
#define LDW    520   // padded LDS row stride (halves) -> conflict-free B reads

typedef _Float16 half_t;
typedef __attribute__((ext_vector_type(16))) _Float16 v16h;
typedef __attribute__((ext_vector_type(8)))  _Float16 v8h;
typedef __attribute__((ext_vector_type(8)))  float    v8f;

#define WMMA_F16(A, B, C) \
  __builtin_amdgcn_wmma_f32_16x16x32_f16(false, (A), false, (B), (short)0, (C), false, false)

// ---- CDNA5 async global->LDS copy (signature learned from round-2 probe:
//      arg0 is a generic `int4*`, arg1 the LDS side, arg2/3 imm offset/cpol) ----
#if defined(__gfx1250__) && __has_builtin(__builtin_amdgcn_global_load_async_to_lds_b128)
#define HAVE_ASYNC_LDS 1
typedef int i32x4 __attribute__((ext_vector_type(4)));
typedef __attribute__((address_space(3))) i32x4 lds_i32x4;

__device__ __forceinline__ void async_copy_b128(const void* gsrc, void* ldst) {
  __builtin_amdgcn_global_load_async_to_lds_b128(
      (i32x4*)(unsigned long long)gsrc,
      (lds_i32x4*)(unsigned)(unsigned long long)ldst,
      0, 0);
}
#endif

// ---------------- WMMA tile loaders (ISA 7.12.2 layouts) ----------------
// A-matrix 16x32 f16: lanes 0-15 hold row M=lane, K = {0..7, 16..23};
//                     lanes 16-31 hold row M=lane-16, K = {8..15, 24..31}.
__device__ __forceinline__ v16h load_a16(const half_t* __restrict__ base, int ld,
                                         int m0, int k0, int lane) {
  const int r  = m0 + (lane & 15);
  const int kb = k0 + ((lane >> 4) << 3);           // +0 or +8
  const half_t* p = base + r * ld + kb;
  v8h lo = *(const v8h*)(p);                        // K = kb .. kb+7
  v8h hi = *(const v8h*)(p + 16);                   // K = kb+16 .. kb+23
  v16h out;
#pragma unroll
  for (int i = 0; i < 8; ++i) { out[i] = lo[i]; out[i + 8] = hi[i]; }
  return out;
}

// B-matrix 32x16 f16: lanes 0-15 hold column N=lane,   K = {0..7, 16..23};
//                     lanes 16-31 hold column N=lane-16, K = {8..15, 24..31}.
// `base` row n supplies COLUMN n of B (x @ W^T / Q @ K^T form).
__device__ __forceinline__ v16h load_b16(const half_t* __restrict__ base, int ld,
                                         int n0, int k0, int lane) {
  const int n  = n0 + (lane & 15);
  const int kb = k0 + ((lane >> 4) << 3);           // +0 or +8
  const half_t* p = base + n * ld + kb;
  v8h lo = *(const v8h*)(p);
  v8h hi = *(const v8h*)(p + 16);
  v16h out;
#pragma unroll
  for (int i = 0; i < 8; ++i) { out[i] = lo[i]; out[i + 8] = hi[i]; }
  return out;
}

// Row-wise reductions inside the 16-lane halves of a C-matrix (wave32).
__device__ __forceinline__ float rmax16(float v) {
#pragma unroll
  for (int m = 8; m >= 1; m >>= 1) v = fmaxf(v, __shfl_xor(v, m, 32));
  return v;
}
__device__ __forceinline__ float rsum16(float v) {
#pragma unroll
  for (int m = 8; m >= 1; m >>= 1) v += __shfl_xor(v, m, 32);
  return v;
}

// Block-cooperative stage of a 64-row x 512-col f16 weight band into padded LDS.
__device__ __forceinline__ void stage_weight_band(half_t* __restrict__ smem,
                                                  const half_t* __restrict__ w,
                                                  int n0) {
  const int t = threadIdx.x;
#pragma unroll
  for (int i = 0; i < 16; ++i) {
    const int idx = i * 256 + t;          // 0..4095 chunks of 8 halves
    const int r   = idx >> 6;             // 0..63
    const int c   = (idx & 63) << 3;      // 0..511 step 8
#if defined(HAVE_ASYNC_LDS)
    async_copy_b128(w + (size_t)(n0 + r) * DMODEL + c, &smem[r * LDW + c]);
#else
    *(v8h*)&smem[r * LDW + c] = *(const v8h*)(w + (size_t)(n0 + r) * DMODEL + c);
#endif
  }
#if defined(HAVE_ASYNC_LDS)
  asm volatile("s_wait_asynccnt 0" ::: "memory");
#endif
  __syncthreads();
}

// ---------------- kernel 1: LayerNorm + cast to f16 ----------------
__global__ void ln_cast_kernel(const float* __restrict__ in,
                               const float* __restrict__ g,
                               const float* __restrict__ b,
                               half_t* __restrict__ xh) {
  const int row  = blockIdx.x;
  const int tid  = threadIdx.x;          // 256 threads, 2 elems each
  const int lane = tid & 31;
  const int wid  = tid >> 5;
  const float* p = in + row * DMODEL;
  const float v0 = p[tid];
  const float v1 = p[tid + 256];
  float s  = v0 + v1;
  float s2 = v0 * v0 + v1 * v1;
#pragma unroll
  for (int m = 16; m >= 1; m >>= 1) {
    s  += __shfl_xor(s,  m, 32);
    s2 += __shfl_xor(s2, m, 32);
  }
  __shared__ float red[16];
  if (lane == 0) { red[wid] = s; red[8 + wid] = s2; }
  __syncthreads();
  float ts = 0.f, tq = 0.f;
#pragma unroll
  for (int i = 0; i < 8; ++i) { ts += red[i]; tq += red[8 + i]; }
  const float mean = ts * (1.0f / DMODEL);
  const float var  = tq * (1.0f / DMODEL) - mean * mean;
  const float rstd = rsqrtf(var + LN_EPS);
  xh[row * DMODEL + tid]       = (half_t)((v0 - mean) * rstd * g[tid]       + b[tid]);
  xh[row * DMODEL + tid + 256] = (half_t)((v1 - mean) * rstd * g[tid + 256] + b[tid + 256]);
}

// ---------------- kernel 2: cast 4 weight matrices to f16 ----------------
__global__ void cast4_kernel(const float* __restrict__ a, const float* __restrict__ b,
                             const float* __restrict__ c, const float* __restrict__ d,
                             half_t* __restrict__ A, half_t* __restrict__ B,
                             half_t* __restrict__ C, half_t* __restrict__ D, int n) {
  const int i = blockIdx.x * blockDim.x + threadIdx.x;
  if (i < n) {
    A[i] = (half_t)a[i]; B[i] = (half_t)b[i];
    C[i] = (half_t)c[i]; D[i] = (half_t)d[i];
  }
}

// ---------------- kernel 3: QKV projections (x @ W^T) ----------------
// grid (32, 8, 3), block 256 (8 waves); weight band staged in LDS once,
// B tiles then come from LDS; A tiles register double-buffered.
__global__ void qkv_gemm_kernel(const half_t* __restrict__ x,
                                const half_t* __restrict__ wq,
                                const half_t* __restrict__ wk,
                                const half_t* __restrict__ wv,
                                half_t* __restrict__ q,
                                half_t* __restrict__ k,
                                half_t* __restrict__ vT) {
  const int lane  = threadIdx.x & 31;
  const int wid   = threadIdx.x >> 5;
  const int m0    = (blockIdx.x * 8 + wid) * 16;
  const int n0    = blockIdx.y * 64;
  const int which = blockIdx.z;
  const half_t* w = (which == 0) ? wq : (which == 1) ? wk : wv;

  __shared__ half_t wsm[64 * LDW];
  stage_weight_band(wsm, w, n0);

  v8f c0 = {}, c1 = {}, c2 = {}, c3 = {};
  v16h a_cur = load_a16(x, DMODEL, m0, 0, lane);
#pragma unroll 4
  for (int k0 = 0; k0 < DMODEL; k0 += 32) {
    v16h a_nxt = a_cur;
    if (k0 + 32 < DMODEL)                 // prefetch next A tile (global)
      a_nxt = load_a16(x, DMODEL, m0, k0 + 32, lane);
    v16h b0 = load_b16(wsm, LDW, 0,  k0, lane);   // B tiles from LDS
    v16h b1 = load_b16(wsm, LDW, 16, k0, lane);
    v16h b2 = load_b16(wsm, LDW, 32, k0, lane);
    v16h b3 = load_b16(wsm, LDW, 48, k0, lane);
    c0 = WMMA_F16(a_cur, b0, c0);
    c1 = WMMA_F16(a_cur, b1, c1);
    c2 = WMMA_F16(a_cur, b2, c2);
    c3 = WMMA_F16(a_cur, b3, c3);
    a_cur = a_nxt;
  }
  const int rb = (lane >> 4) << 3;
  const int nc = lane & 15;
  if (which == 2) {                       // V transposed: vT[n][m]
#pragma unroll
    for (int r = 0; r < 8; ++r) {
      const int m = m0 + r + rb;
      vT[(n0      + nc) * BDIM + m] = (half_t)c0[r];
      vT[(n0 + 16 + nc) * BDIM + m] = (half_t)c1[r];
      vT[(n0 + 32 + nc) * BDIM + m] = (half_t)c2[r];
      vT[(n0 + 48 + nc) * BDIM + m] = (half_t)c3[r];
    }
  } else {
    half_t* dst = (which == 0) ? q : k;
#pragma unroll
    for (int r = 0; r < 8; ++r) {
      const int m = m0 + r + rb;
      dst[m * DMODEL + n0      + nc] = (half_t)c0[r];
      dst[m * DMODEL + n0 + 16 + nc] = (half_t)c1[r];
      dst[m * DMODEL + n0 + 32 + nc] = (half_t)c2[r];
      dst[m * DMODEL + n0 + 48 + nc] = (half_t)c3[r];
    }
  }
}

// ---------------- kernel 4: flash attention (leaky-relu scores) ----------------
// block = 256 (8 waves); each wave owns one (head, 16-row query band).
// K-chunk B-tiles double-buffered; V B-tiles hoisted above softmax so their
// latency overlaps the VALU-heavy online-softmax section.
__global__ void flash_attn_kernel(const half_t* __restrict__ qg,
                                  const half_t* __restrict__ kg,
                                  const half_t* __restrict__ vT,
                                  half_t* __restrict__ ao) {
  const int lane = threadIdx.x & 31;
  const int wid  = threadIdx.x >> 5;
  const int gid  = blockIdx.x * 8 + wid;     // 0..2047
  const int band = gid & 255;                // query band (16 rows)
  const int h    = gid >> 8;                 // head

  const half_t* qh  = qg + h * DHEAD;        // row-major, ld = DMODEL
  const half_t* kh  = kg + h * DHEAD;
  const half_t* vh  = vT + (size_t)h * DHEAD * BDIM; // [DHEAD][BDIM]

  __shared__ half_t plds_all[8][16 * 32];    // per-wave P staging tile
  half_t* plds = plds_all[wid];

  const float scale = 0.044194173824159216f; // 1/sqrt(512)

  const v16h aq0 = load_a16(qh, DMODEL, band * 16, 0,  lane);
  const v16h aq1 = load_a16(qh, DMODEL, band * 16, 32, lane);

  v8f o0 = {}, o1 = {}, o2 = {}, o3 = {};
  float mrun[8], lrun[8];
#pragma unroll
  for (int r = 0; r < 8; ++r) { mrun[r] = -1e30f; lrun[r] = 0.0f; }

  const int rb = (lane >> 4) << 3;
  const int nc = lane & 15;

  // prologue: first K-chunk B tiles
  v16h bk00 = load_b16(kh, DMODEL, 0,  0,  lane);
  v16h bk01 = load_b16(kh, DMODEL, 0,  32, lane);
  v16h bk10 = load_b16(kh, DMODEL, 16, 0,  lane);
  v16h bk11 = load_b16(kh, DMODEL, 16, 32, lane);

  for (int j0 = 0; j0 < BDIM; j0 += 32) {
    // issue NEXT chunk's K tiles early (register double buffer)
    v16h nk00 = bk00, nk01 = bk01, nk10 = bk10, nk11 = bk11;
    if (j0 + 32 < BDIM) {
      nk00 = load_b16(kh, DMODEL, j0 + 32, 0,  lane);
      nk01 = load_b16(kh, DMODEL, j0 + 32, 32, lane);
      nk10 = load_b16(kh, DMODEL, j0 + 48, 0,  lane);
      nk11 = load_b16(kh, DMODEL, j0 + 48, 32, lane);
      __builtin_prefetch(kh + (size_t)(j0 + 64 + nc) * DMODEL, 0, 0);
    }
    // issue V tiles for THIS chunk early: overlap with softmax VALU below
    v16h bv0 = load_b16(vh, BDIM, 0,  j0, lane);
    v16h bv1 = load_b16(vh, BDIM, 16, j0, lane);
    v16h bv2 = load_b16(vh, BDIM, 32, j0, lane);
    v16h bv3 = load_b16(vh, BDIM, 48, j0, lane);

    // ---- scores S = (Q K^T) for 32 keys: two 16x16 tiles ----
    v8f s0 = {}, s1 = {};
    s0 = WMMA_F16(aq0, bk00, s0);
    s0 = WMMA_F16(aq1, bk01, s0);
    s1 = WMMA_F16(aq0, bk10, s1);
    s1 = WMMA_F16(aq1, bk11, s1);

    // ---- leaky-relu + online softmax (rows live across 16-lane halves) ----
#pragma unroll
    for (int r = 0; r < 8; ++r) {
      float x0 = s0[r] * scale; x0 = (x0 > 0.0f) ? x0 : NEG_SLOPE * x0;
      float x1 = s1[r] * scale; x1 = (x1 > 0.0f) ? x1 : NEG_SLOPE * x1;
      const float mx   = rmax16(fmaxf(x0, x1));
      const float mnew = fmaxf(mrun[r], mx);
      const float fac  = __expf(mrun[r] - mnew);
      const float p0   = __expf(x0 - mnew);
      const float p1   = __expf(x1 - mnew);
      lrun[r] = lrun[r] * fac + rsum16(p0 + p1);
      mrun[r] = mnew;
      o0[r] *= fac; o1[r] *= fac; o2[r] *= fac; o3[r] *= fac;
      const int row = r + rb;                 // C-layout -> LDS row-major P
      plds[row * 32 + nc]      = (half_t)p0;
      plds[row * 32 + 16 + nc] = (half_t)p1;
    }
    asm volatile("s_wait_dscnt 0" ::: "memory");  // LDS stores visible to loads

    // ---- O += P (16x32) @ V_chunk (32x64) ----
    const v16h pa = load_a16(plds, 32, 0, 0, lane);
    o0 = WMMA_F16(pa, bv0, o0);
    o1 = WMMA_F16(pa, bv1, o1);
    o2 = WMMA_F16(pa, bv2, o2);
    o3 = WMMA_F16(pa, bv3, o3);

    bk00 = nk00; bk01 = nk01; bk10 = nk10; bk11 = nk11;
  }

  // ---- normalize and write f16 activation for the FC GEMM ----
#pragma unroll
  for (int r = 0; r < 8; ++r) {
    const float inv = 1.0f / lrun[r];
    const int row  = band * 16 + r + rb;
    const int colb = h * DHEAD + nc;
    ao[row * DMODEL + colb]      = (half_t)(o0[r] * inv);
    ao[row * DMODEL + colb + 16] = (half_t)(o1[r] * inv);
    ao[row * DMODEL + colb + 32] = (half_t)(o2[r] * inv);
    ao[row * DMODEL + colb + 48] = (half_t)(o3[r] * inv);
  }
}

// ---------------- kernel 5: FC (ao @ fc_w^T) + bias + residual ----------------
__global__ void fc_gemm_kernel(const half_t* __restrict__ ao,
                               const half_t* __restrict__ w,
                               const float* __restrict__ bias,
                               const float* __restrict__ resid,
                               float* __restrict__ out) {
  const int lane = threadIdx.x & 31;
  const int wid  = threadIdx.x >> 5;
  const int m0   = (blockIdx.x * 8 + wid) * 16;
  const int n0   = blockIdx.y * 64;

  __shared__ half_t wsm[64 * LDW];
  stage_weight_band(wsm, w, n0);

  v8f c0 = {}, c1 = {}, c2 = {}, c3 = {};
  v16h a_cur = load_a16(ao, DMODEL, m0, 0, lane);
#pragma unroll 4
  for (int k0 = 0; k0 < DMODEL; k0 += 32) {
    v16h a_nxt = a_cur;
    if (k0 + 32 < DMODEL)
      a_nxt = load_a16(ao, DMODEL, m0, k0 + 32, lane);
    v16h b0 = load_b16(wsm, LDW, 0,  k0, lane);
    v16h b1 = load_b16(wsm, LDW, 16, k0, lane);
    v16h b2 = load_b16(wsm, LDW, 32, k0, lane);
    v16h b3 = load_b16(wsm, LDW, 48, k0, lane);
    c0 = WMMA_F16(a_cur, b0, c0);
    c1 = WMMA_F16(a_cur, b1, c1);
    c2 = WMMA_F16(a_cur, b2, c2);
    c3 = WMMA_F16(a_cur, b3, c3);
    a_cur = a_nxt;
  }
  const int rb = (lane >> 4) << 3;
  const int nc = lane & 15;
#pragma unroll
  for (int r = 0; r < 8; ++r) {
    const int m = m0 + r + rb;
    out[m * DMODEL + n0      + nc] = c0[r] + bias[n0      + nc] + resid[m * DMODEL + n0      + nc];
    out[m * DMODEL + n0 + 16 + nc] = c1[r] + bias[n0 + 16 + nc] + resid[m * DMODEL + n0 + 16 + nc];
    out[m * DMODEL + n0 + 32 + nc] = c2[r] + bias[n0 + 32 + nc] + resid[m * DMODEL + n0 + 32 + nc];
    out[m * DMODEL + n0 + 48 + nc] = c3[r] + bias[n0 + 48 + nc] + resid[m * DMODEL + n0 + 48 + nc];
  }
}

// ---------------- host launcher ----------------
extern "C" void kernel_launch(void* const* d_in, const int* in_sizes, int n_in,
                              void* d_out, int out_size, void* d_ws, size_t ws_size,
                              hipStream_t stream) {
  const float* in_feats = (const float*)d_in[0];
  const float* wq   = (const float*)d_in[1];
  const float* wk   = (const float*)d_in[2];
  const float* wv   = (const float*)d_in[3];
  const float* fc_w = (const float*)d_in[4];
  const float* fc_b = (const float*)d_in[5];
  const float* ln_w = (const float*)d_in[6];
  const float* ln_b = (const float*)d_in[7];
  float* out = (float*)d_out;

  // workspace carve-up (f16): 23 MB total
  half_t* xh  = (half_t*)d_ws;                 // [B][D] normalized input
  half_t* wqh = xh  + (size_t)BDIM * DMODEL;
  half_t* wkh = wqh + (size_t)DMODEL * DMODEL;
  half_t* wvh = wkh + (size_t)DMODEL * DMODEL;
  half_t* fch = wvh + (size_t)DMODEL * DMODEL;
  half_t* qh  = fch + (size_t)DMODEL * DMODEL; // [B][D]
  half_t* kh  = qh  + (size_t)BDIM * DMODEL;   // [B][D]
  half_t* vT  = kh  + (size_t)BDIM * DMODEL;   // [D][B] (per-head [64][B] slabs)
  half_t* ao  = vT  + (size_t)BDIM * DMODEL;   // [B][D] attention output

  ln_cast_kernel<<<BDIM, 256, 0, stream>>>(in_feats, ln_w, ln_b, xh);
  cast4_kernel<<<(DMODEL * DMODEL + 255) / 256, 256, 0, stream>>>(
      wq, wk, wv, fc_w, wqh, wkh, wvh, fch, DMODEL * DMODEL);
  qkv_gemm_kernel<<<dim3(BDIM / 128, DMODEL / 64, 3), 256, 0, stream>>>(
      xh, wqh, wkh, wvh, qh, kh, vT);
  flash_attn_kernel<<<(NHEADS * (BDIM / 16)) / 8, 256, 0, stream>>>(qh, kh, vT, ao);
  fc_gemm_kernel<<<dim3(BDIM / 128, DMODEL / 64), 256, 0, stream>>>(
      ao, fch, fc_b, in_feats, out);
}
